// GraphTransformer_32263794328105
// MI455X (gfx1250) — matrix operand order repaired and checked by reading, not compiled, not run on hardware
//
#include <hip/hip_runtime.h>
#include <hip/hip_bf16.h>

#define NUM_RBF 16
#define K_NB 30
#define N_LAYERS 4
#define N_HEADS 4
#define HID 64
#define B_ 8
#define L_ 2048
#define F_ 1038
#define BL (B_*L_)
#define NE (BL*K_NB)

typedef __attribute__((ext_vector_type(16))) _Float16 v16h;
typedef __attribute__((ext_vector_type(8)))  float    v8f;

__device__ inline v8f wmma_f16(v16h a, v16h b, v8f c) {
  return __builtin_amdgcn_wmma_f32_16x16x32_f16(false, a, false, b, (short)0, c, false, false);
}

// ---------------------------------------------------------------------------
// Rearrange a row-major f32 weight (K x N) into WMMA B-fragment layout:
// tile (kt, nt): lane holds column n = nt*16 + (lane&15); halves e=0..15 hold
// k = kt*32 + 8*(lane>>4) + e (e<8), kt*32 + 16 + 8*(lane>>4) + (e-8) (e>=8).   One block (32 threads) per tile.
// ---------------------------------------------------------------------------
__global__ void frag_weights_kernel(const float* __restrict__ W, _Float16* __restrict__ frag,
                                    int K, int N, int ntiles) {
  int kt = blockIdx.x / ntiles;
  int nt = blockIdx.x - kt*ntiles;
  int lane = threadIdx.x;
  int n = nt*16 + (lane & 15);
  int hi = lane >> 4;
  _Float16* dst = frag + ((size_t)(kt*ntiles + nt)*32 + lane)*16;
#pragma unroll
  for (int e = 0; e < 16; ++e) {
    int k = kt*32 + (e < 8 ? 8*hi + e : 16 + 8*hi + (e - 8));
    float v = (k < K) ? W[(size_t)k*N + n] : 0.0f;
    dst[e] = (_Float16)v;
  }
}

// ---------------------------------------------------------------------------
// KNN: one block per (b,i). Computes distances to all L nodes, row max,
// masked-adjusted distances, then iteratively extracts the 30 smallest using
// packed (float_bits<<32 | index) u64 keys (ties -> lowest index, like top_k).
// ---------------------------------------------------------------------------
__global__ void knn_kernel(const float* __restrict__ coords, const float* __restrict__ mask,
                           int* __restrict__ E_idx, float* __restrict__ D_nb) {
  __shared__ float sD[L_];
  __shared__ unsigned long long sKey[L_];
  __shared__ unsigned long long sRedK[256];
  __shared__ float sRedF[256];
  __shared__ float sDmax;
  int bi = blockIdx.x;
  int b = bi / L_;
  int tid = threadIdx.x;
  float cx = coords[(size_t)bi*3+0];
  float cy = coords[(size_t)bi*3+1];
  float cz = coords[(size_t)bi*3+2];
  float mi = mask[bi];
  float lmax = 0.0f;
  for (int j = tid; j < L_; j += 256) {
    size_t cj = (size_t)(b*L_ + j)*3;
    float dx = cx - coords[cj], dy = cy - coords[cj+1], dz = cz - coords[cj+2];
    float D = sqrtf(dx*dx + dy*dy + dz*dz + 1e-6f);
    sD[j] = D;
    lmax = fmaxf(lmax, D);
  }
  sRedF[tid] = lmax;
  __syncthreads();
  for (int s = 128; s > 0; s >>= 1) {
    if (tid < s) sRedF[tid] = fmaxf(sRedF[tid], sRedF[tid+s]);
    __syncthreads();
  }
  if (tid == 0) sDmax = sRedF[0];
  __syncthreads();
  float Dmax = sDmax;
  for (int j = tid; j < L_; j += 256) {
    float mj = mask[b*L_ + j];
    float Dadj = sD[j] + (1.0f - mi*mj) * Dmax;
    sKey[j] = ((unsigned long long)__float_as_uint(Dadj) << 32) | (unsigned)j;
  }
  __syncthreads();
  for (int t = 0; t < K_NB; ++t) {
    unsigned long long mn = ~0ull;
    for (int j = tid; j < L_; j += 256) {
      unsigned long long k = sKey[j];
      mn = (k < mn) ? k : mn;
    }
    sRedK[tid] = mn;
    __syncthreads();
    for (int s = 128; s > 0; s >>= 1) {
      if (tid < s) {
        unsigned long long a = sRedK[tid], c = sRedK[tid+s];
        sRedK[tid] = (c < a) ? c : a;
      }
      __syncthreads();
    }
    if (tid == 0) {
      unsigned long long k = sRedK[0];
      int j = (int)(k & 0xffffffffull);
      E_idx[(size_t)bi*K_NB + t] = j;
      D_nb[(size_t)bi*K_NB + t] = __uint_as_float((unsigned)(k >> 32));
      sKey[j] = ~0ull;
    }
    __syncthreads();
  }
}

// ---------------------------------------------------------------------------
// Edge embedding: per edge-row thread: RBF(16) -> @W_ee -> LN -> @W_ep + b_ep
// Output h_E as f16 rows (A-matrix source for K/V GEMMs).
// ---------------------------------------------------------------------------
__global__ void edge_embed_kernel(const float* __restrict__ D_nb,
                                  const float* __restrict__ W_ee,
                                  const float* __restrict__ g_eln, const float* __restrict__ b_eln,
                                  const float* __restrict__ W_ep, const float* __restrict__ b_ep,
                                  _Float16* __restrict__ hE) {
  __shared__ float sWee[256];
  __shared__ float sWep[1024];
  __shared__ float sG[16], sB[16], sBep[64];
  int tid = threadIdx.x;
  sWee[tid] = W_ee[tid];
  for (int t = tid; t < 1024; t += 256) sWep[t] = W_ep[t];
  if (tid < 16) { sG[tid] = g_eln[tid]; sB[tid] = b_eln[tid]; }
  if (tid < 64) sBep[tid] = b_ep[tid];
  __syncthreads();
  size_t e = (size_t)blockIdx.x*256 + tid;
  if (e >= (size_t)NE) return;
  float D = D_nb[e];
  float rbf[16];
#pragma unroll
  for (int r = 0; r < 16; ++r) {
    float mu = 2.0f + (20.0f/15.0f)*(float)r;   // linspace(2,22,16)
    float t2 = (D - mu) * (1.0f/1.25f);          // sigma = 20/16
    rbf[r] = __expf(-t2*t2);
  }
  float E0[16];
  float mean = 0.0f;
#pragma unroll
  for (int c = 0; c < 16; ++c) {
    float s = 0.0f;
#pragma unroll
    for (int r = 0; r < 16; ++r) s += rbf[r]*sWee[r*16 + c];
    E0[c] = s; mean += s;
  }
  mean *= (1.0f/16.0f);
  float var = 0.0f;
#pragma unroll
  for (int c = 0; c < 16; ++c) { float d = E0[c]-mean; var += d*d; }
  float inv = rsqrtf(var*(1.0f/16.0f) + 1e-5f);
  float En[16];
#pragma unroll
  for (int c = 0; c < 16; ++c) En[c] = (E0[c]-mean)*inv*sG[c] + sB[c];
  _Float16* orow = hE + e*64;
#pragma unroll 4
  for (int n = 0; n < 64; ++n) {
    float s = sBep[n];
#pragma unroll
    for (int c = 0; c < 16; ++c) s += En[c]*sWep[c*64 + n];
    orow[n] = (_Float16)s;
  }
}

// ---------------------------------------------------------------------------
// Generic WMMA GEMM: one wave per 16x64 tile. A row-major (f32 or f16, lda=K),
// B in fragment layout, bias per column. EPI: 0=f32, 1=f16, 2=f32+f16, 3=relu f16
// ---------------------------------------------------------------------------
template<typename AT>
__device__ inline v16h load_a_frag(const AT* __restrict__ p, int k0, int K, int hi) {
  v16h f;
  if (k0 + 32 <= K) {
#pragma unroll
    for (int e = 0; e < 8; ++e) f[e]   = (_Float16)p[k0 + 8*hi + e];
#pragma unroll
    for (int e = 0; e < 8; ++e) f[8+e] = (_Float16)p[k0 + 16 + 8*hi + e];
  } else {
#pragma unroll
    for (int e = 0; e < 8; ++e) { int k = k0 + 8*hi + e;      f[e]   = (k < K) ? (_Float16)p[k] : (_Float16)0.0f; }
#pragma unroll
    for (int e = 0; e < 8; ++e) { int k = k0 + 16 + 8*hi + e; f[8+e] = (k < K) ? (_Float16)p[k] : (_Float16)0.0f; }
  }
  return f;
}

template<typename AT, int EPI>
__global__ void gemm_wmma_kernel(const AT* __restrict__ A, const _Float16* __restrict__ Bfrag,
                                 const float* __restrict__ bias,
                                 float* __restrict__ Cf, _Float16* __restrict__ Ch,
                                 int M, int N, int K) {
  int lane = threadIdx.x & 31;
  int wid = blockIdx.x * (blockDim.x >> 5) + (threadIdx.x >> 5);
  int nchunks = N >> 6;
  int total = (M >> 4) * nchunks;
  if (wid >= total) return;
  int mt = wid / nchunks, nc = wid - mt*nchunks;
  int ntiles = N >> 4;
  int m = lane & 15, hi = lane >> 4;
  const AT* arow = A + (size_t)(mt*16 + m) * (size_t)K;
  v8f acc[4] = {};
  int ktiles = (K + 31) >> 5;
  const v16h* Bf = (const v16h*)Bfrag;
  for (int kt = 0; kt < ktiles; ++kt) {
    v16h a = load_a_frag<AT>(arow, kt*32, K, hi);
#pragma unroll
    for (int t = 0; t < 4; ++t) {
      v16h b = Bf[(size_t)(kt*ntiles + nc*4 + t)*32 + lane];
      acc[t] = wmma_f16(a, b, acc[t]);
    }
  }
#pragma unroll
  for (int t = 0; t < 4; ++t) {
    int n = nc*64 + t*16 + m;
    float bv = bias ? bias[n] : 0.0f;
#pragma unroll
    for (int v = 0; v < 8; ++v) {
      int row = mt*16 + v + 8*hi;
      float val = acc[t][v] + bv;
      if constexpr (EPI == 3) val = fmaxf(val, 0.0f);
      size_t o = (size_t)row*(size_t)N + n;
      if constexpr (EPI == 0 || EPI == 2) Cf[o] = val;
      if constexpr (EPI == 1 || EPI == 2 || EPI == 3) Ch[o] = (_Float16)val;
    }
  }
}

// ---------------------------------------------------------------------------
// Fused K/V GEMM over h_EV = concat(h_E, gather(h_V)).  One wave per 16 edge
// rows; gather is folded into the per-lane A-fragment load; A fragments are
// shared between the K and V projections (K=128, N=64 each).
// ---------------------------------------------------------------------------
__global__ void kv_gemm_kernel(const _Float16* __restrict__ hE, const _Float16* __restrict__ hV16,
                               const int* __restrict__ E_idx,
                               const _Float16* __restrict__ BfK, const _Float16* __restrict__ BfV,
                               const float* __restrict__ biasK, const float* __restrict__ biasV,
                               _Float16* __restrict__ Kt, _Float16* __restrict__ Vt) {
  int lane = threadIdx.x & 31;
  int wid = blockIdx.x * (blockDim.x >> 5) + (threadIdx.x >> 5);
  int mtiles = NE >> 4;
  if (wid >= mtiles) return;
  int m = lane & 15, hi = lane >> 4;
  int e = wid*16 + m;
  int b = e / (L_*K_NB);
  int gi = E_idx[e];
  const _Float16* pE = hE + (size_t)e*64;
  const _Float16* pV = hV16 + (size_t)(b*L_ + gi)*64;
  v8f accK[4] = {};
  v8f accV[4] = {};
  const v16h* BK = (const v16h*)BfK;
  const v16h* BV = (const v16h*)BfV;
#pragma unroll
  for (int kt = 0; kt < 4; ++kt) {
    const _Float16* p = (kt < 2) ? pE : pV;
    int k0 = (kt & 1)*32;
    v16h a;
#pragma unroll
    for (int q = 0; q < 8; ++q) a[q]   = p[k0 + 8*hi + q];
#pragma unroll
    for (int q = 0; q < 8; ++q) a[8+q] = p[k0 + 16 + 8*hi + q];
#pragma unroll
    for (int t = 0; t < 4; ++t) {
      accK[t] = wmma_f16(a, BK[(size_t)(kt*4 + t)*32 + lane], accK[t]);
      accV[t] = wmma_f16(a, BV[(size_t)(kt*4 + t)*32 + lane], accV[t]);
    }
  }
#pragma unroll
  for (int t = 0; t < 4; ++t) {
    int n = t*16 + m;
    float bk = biasK[n], bv = biasV[n];
#pragma unroll
    for (int v = 0; v < 8; ++v) {
      size_t row = (size_t)wid*16 + v + 8*hi;
      Kt[row*64 + n] = (_Float16)(accK[t][v] + bk);
      Vt[row*64 + n] = (_Float16)(accV[t][v] + bv);
    }
  }
}

// ---------------------------------------------------------------------------
// Attention: one wave per (node, head). lane = neighbor index (30 active).
// ---------------------------------------------------------------------------
__global__ void attn_kernel(const _Float16* __restrict__ Q, const _Float16* __restrict__ Kt,
                            const _Float16* __restrict__ Vt, const float* __restrict__ mask,
                            const int* __restrict__ E_idx, _Float16* __restrict__ upd) {
  int lane = threadIdx.x & 31;
  int wid = blockIdx.x * (blockDim.x >> 5) + (threadIdx.x >> 5);
  int row = wid >> 2;
  int h = wid & 3;
  if (row >= BL) return;
  int b = row / L_;
  bool act = lane < K_NB;
  int kk = act ? lane : 0;
  size_t e = (size_t)row*K_NB + kk;
  v16h q = *(const v16h*)(Q + (size_t)row*64 + h*16);
  v16h kv = {};
  if (act) kv = *(const v16h*)(Kt + e*64 + h*16);
  float logit = 0.0f;
#pragma unroll
  for (int d = 0; d < 16; ++d) logit += (float)q[d] * (float)kv[d];
  logit *= 0.25f;   // 1/sqrt(16)
  float mk = 0.0f;
  if (act) mk = mask[b*L_ + E_idx[e]] * mask[row];
  if (!(act && mk > 0.0f)) logit = -3.402823466e38f;
  float mx = logit;
#pragma unroll
  for (int s = 16; s > 0; s >>= 1) mx = fmaxf(mx, __shfl_xor(mx, s, 32));
  float ex = act ? __expf(logit - mx) : 0.0f;
  float sum = ex;
#pragma unroll
  for (int s = 16; s > 0; s >>= 1) sum += __shfl_xor(sum, s, 32);
  float attend = act ? (ex / sum) * mk : 0.0f;
  v16h vv = {};
  if (act) vv = *(const v16h*)(Vt + e*64 + h*16);
  float u[16];
#pragma unroll
  for (int d = 0; d < 16; ++d) u[d] = attend * (float)vv[d];
#pragma unroll
  for (int s = 16; s > 0; s >>= 1) {
#pragma unroll
    for (int d = 0; d < 16; ++d) u[d] += __shfl_xor(u[d], s, 32);
  }
  if (lane == 0) {
    v16h o;
#pragma unroll
    for (int d = 0; d < 16; ++d) o[d] = (_Float16)u[d];
    *(v16h*)(upd + (size_t)row*64 + h*16) = o;
  }
}

// ---------------------------------------------------------------------------
// Residual + LayerNorm (+optional mask multiply). One wave per row; writes
// f32 state and the f16 shadow used as the next GEMM's A matrix.
// ---------------------------------------------------------------------------
template<bool MASKMUL>
__global__ void ln_kernel(float* __restrict__ hV, const float* __restrict__ dh,
                          const float* __restrict__ g, const float* __restrict__ bb,
                          const float* __restrict__ mask, _Float16* __restrict__ hV16) {
  int lane = threadIdx.x & 31;
  int row = blockIdx.x * (blockDim.x >> 5) + (threadIdx.x >> 5);
  if (row >= BL) return;
  size_t o = (size_t)row*64;
  int e0 = lane*2, e1 = lane*2 + 1;
  float x0 = hV[o+e0] + dh[o+e0];
  float x1 = hV[o+e1] + dh[o+e1];
  float s = x0 + x1;
#pragma unroll
  for (int t = 16; t > 0; t >>= 1) s += __shfl_xor(s, t, 32);
  float mean = s * (1.0f/64.0f);
  float d0 = x0 - mean, d1 = x1 - mean;
  float vv = d0*d0 + d1*d1;
#pragma unroll
  for (int t = 16; t > 0; t >>= 1) vv += __shfl_xor(vv, t, 32);
  float inv = rsqrtf(vv*(1.0f/64.0f) + 1e-5f);
  float mm = MASKMUL ? mask[row] : 1.0f;
  float y0 = (d0*inv*g[e0] + bb[e0]) * mm;
  float y1 = (d1*inv*g[e1] + bb[e1]) * mm;
  hV[o+e0] = y0; hV[o+e1] = y1;
  hV16[o+e0] = (_Float16)y0; hV16[o+e1] = (_Float16)y1;
}

// ---------------------------------------------------------------------------
// Output head: one wave per row, dot(h_V[row], W_out) + b_out.
// ---------------------------------------------------------------------------
__global__ void out_kernel(const float* __restrict__ hV, const float* __restrict__ Wout,
                           const float* __restrict__ bout, float* __restrict__ out) {
  int lane = threadIdx.x & 31;
  int row = blockIdx.x * (blockDim.x >> 5) + (threadIdx.x >> 5);
  if (row >= BL) return;
  size_t o = (size_t)row*64;
  float s = hV[o+lane]*Wout[lane] + hV[o+32+lane]*Wout[32+lane];
#pragma unroll
  for (int t = 16; t > 0; t >>= 1) s += __shfl_xor(s, t, 32);
  if (lane == 0) out[row] = s + bout[0];
}

// ---------------------------------------------------------------------------
extern "C" void kernel_launch(void* const* d_in, const int* in_sizes, int n_in,
                              void* d_out, int out_size, void* d_ws, size_t ws_size,
                              hipStream_t stream) {
  const float* coords        = (const float*)d_in[0];
  const float* node_features = (const float*)d_in[1];
  const float* mask          = (const float*)d_in[2];
  const float* W_node        = (const float*)d_in[3];
  const float* b_node        = (const float*)d_in[4];
  const float* W_ee          = (const float*)d_in[5];
  const float* g_eln         = (const float*)d_in[6];
  const float* b_eln         = (const float*)d_in[7];
  const float* W_ep          = (const float*)d_in[8];
  const float* b_ep          = (const float*)d_in[9];
  const float* WQ            = (const float*)d_in[10];
  const float* bQ            = (const float*)d_in[11];
  const float* WK            = (const float*)d_in[12];
  const float* bK            = (const float*)d_in[13];
  const float* WV            = (const float*)d_in[14];
  const float* bV            = (const float*)d_in[15];
  const float* WO            = (const float*)d_in[16];
  const float* bO            = (const float*)d_in[17];
  const float* g1            = (const float*)d_in[18];
  const float* b1            = (const float*)d_in[19];
  const float* W1            = (const float*)d_in[20];
  const float* bf1           = (const float*)d_in[21];
  const float* W2            = (const float*)d_in[22];
  const float* bf2           = (const float*)d_in[23];
  const float* g2            = (const float*)d_in[24];
  const float* b2            = (const float*)d_in[25];
  const float* W_out         = (const float*)d_in[26];
  const float* b_out         = (const float*)d_in[27];
  float* out = (float*)d_out;
  (void)in_sizes; (void)n_in; (void)out_size; (void)ws_size;

  char* ws = (char*)d_ws;
  size_t off = 0;
  auto alloc = [&](size_t bytes) -> char* {
    off = (off + 255) & ~(size_t)255;
    char* p = ws + off;
    off += bytes;
    return p;
  };

  int*      eidx  = (int*)alloc((size_t)NE*4);
  float*    dnb   = (float*)alloc((size_t)NE*4);
  _Float16* hE    = (_Float16*)alloc((size_t)NE*64*2);
  _Float16* Kt16  = (_Float16*)alloc((size_t)NE*64*2);
  _Float16* Vt16  = (_Float16*)alloc((size_t)NE*64*2);
  float*    hV    = (float*)alloc((size_t)BL*64*4);
  _Float16* hV16  = (_Float16*)alloc((size_t)BL*64*2);
  _Float16* Q16   = (_Float16*)alloc((size_t)BL*64*2);
  _Float16* upd16 = (_Float16*)alloc((size_t)BL*64*2);
  float*    dh    = (float*)alloc((size_t)BL*64*4);
  _Float16* ffn1  = (_Float16*)alloc((size_t)BL*256*2);

  _Float16* fWnode = (_Float16*)alloc((size_t)33*4*512*2);
  _Float16 *fWQ[4], *fWK[4], *fWV[4], *fWO[4], *fW1[4], *fW2[4];
  for (int l = 0; l < N_LAYERS; ++l) {
    fWQ[l] = (_Float16*)alloc((size_t)2*4*512*2);
    fWK[l] = (_Float16*)alloc((size_t)4*4*512*2);
    fWV[l] = (_Float16*)alloc((size_t)4*4*512*2);
    fWO[l] = (_Float16*)alloc((size_t)2*4*512*2);
    fW1[l] = (_Float16*)alloc((size_t)2*16*512*2);
    fW2[l] = (_Float16*)alloc((size_t)8*4*512*2);
  }

  // --- weight fragment prep ---
  frag_weights_kernel<<<33*4, 32, 0, stream>>>(W_node, fWnode, F_, 64, 4);
  for (int l = 0; l < N_LAYERS; ++l) {
    frag_weights_kernel<<<2*4,  32, 0, stream>>>(WQ + (size_t)l*64*64,  fWQ[l], 64,  64,  4);
    frag_weights_kernel<<<4*4,  32, 0, stream>>>(WK + (size_t)l*128*64, fWK[l], 128, 64,  4);
    frag_weights_kernel<<<4*4,  32, 0, stream>>>(WV + (size_t)l*128*64, fWV[l], 128, 64,  4);
    frag_weights_kernel<<<2*4,  32, 0, stream>>>(WO + (size_t)l*64*64,  fWO[l], 64,  64,  4);
    frag_weights_kernel<<<2*16, 32, 0, stream>>>(W1 + (size_t)l*64*256, fW1[l], 64,  256, 16);
    frag_weights_kernel<<<8*4,  32, 0, stream>>>(W2 + (size_t)l*256*64, fW2[l], 256, 64,  4);
  }

  // --- graph construction + edge embedding ---
  knn_kernel<<<BL, 256, 0, stream>>>(coords, mask, eidx, dnb);
  edge_embed_kernel<<<(NE + 255)/256, 256, 0, stream>>>(dnb, W_ee, g_eln, b_eln, W_ep, b_ep, hE);

  // --- node embedding: h_V = nf @ W_node + b (writes f32 + f16 shadow) ---
  gemm_wmma_kernel<float, 2><<<256, 128, 0, stream>>>(node_features, fWnode, b_node,
                                                      hV, hV16, BL, 64, F_);

  // --- transformer layers ---
  for (int l = 0; l < N_LAYERS; ++l) {
    gemm_wmma_kernel<_Float16, 1><<<256, 128, 0, stream>>>(hV16, fWQ[l], bQ + l*64,
                                                           nullptr, Q16, BL, 64, 64);
    kv_gemm_kernel<<<(NE/16)/4, 128, 0, stream>>>(hE, hV16, eidx, fWK[l], fWV[l],
                                                  bK + l*64, bV + l*64, Kt16, Vt16);
    attn_kernel<<<(BL*N_HEADS)/8, 256, 0, stream>>>(Q16, Kt16, Vt16, mask, eidx, upd16);
    gemm_wmma_kernel<_Float16, 0><<<256, 128, 0, stream>>>(upd16, fWO[l], bO + l*64,
                                                           dh, nullptr, BL, 64, 64);
    ln_kernel<false><<<BL/8, 256, 0, stream>>>(hV, dh, g1 + l*64, b1 + l*64, mask, hV16);
    gemm_wmma_kernel<_Float16, 3><<<1024, 128, 0, stream>>>(hV16, fW1[l], bf1 + l*256,
                                                            nullptr, ffn1, BL, 256, 64);
    gemm_wmma_kernel<_Float16, 0><<<256, 128, 0, stream>>>(ffn1, fW2[l], bf2 + l*64,
                                                           dh, nullptr, BL, 64, 256);
    ln_kernel<true><<<BL/8, 256, 0, stream>>>(hV, dh, g2 + l*64, b2 + l*64, mask, hV16);
  }

  out_kernel<<<BL/8, 256, 0, stream>>>(hV, W_out, b_out, out);
}
